// S2SBiLSTM_26834955665876
// MI455X (gfx1250) — matrix operand order repaired and checked by reading, not compiled, not run on hardware
//
#include <hip/hip_runtime.h>

// ---------------------------------------------------------------------------
// Seq2Seq BiLSTM for MI455X (gfx1250, wave32, WMMA, async-to-LDS staging).
// V=32000 E=256 H=512 B=32 S=128 T=64
//   * bf16 WMMA (v_wmma_f32_16x16x32_bf16) everywhere, fp32 accumulate.
//   * Input-side LSTM GEMMs hoisted out of the recurrences.
//   * Fused per-step LSTM kernel (h@Whh^T + gates + cell update), direct loads
//     (latency-bound serial chain; Whh is L2-resident).
//   * Block GEMM: 512 thr = 4x4 waves on a 64x256 macro-tile, operands staged
//     into LDS with GLOBAL_LOAD_ASYNC_TO_LDS_B128 (ASYNCcnt), double-buffered.
// ---------------------------------------------------------------------------

typedef __bf16 bf16_t;
typedef __attribute__((ext_vector_type(16))) bf16_t v16bf;
typedef __attribute__((ext_vector_type(8)))  bf16_t v8bf;
typedef __attribute__((ext_vector_type(8)))  float  v8f;

// Pointer types for the async global->LDS builtin (param 0 is int4* in AS1
// per the compiler diagnostic; param 1 assumed int4* in AS3).
typedef int v4i_t __attribute__((vector_size(16)));
typedef __attribute__((address_space(1))) void*   as1_vp;
typedef __attribute__((address_space(3))) void*   as3_vp;
typedef __attribute__((address_space(1))) v4i_t*  as1_v4ip;
typedef __attribute__((address_space(3))) v4i_t*  as3_v4ip;

#if defined(__has_builtin)
# if __has_builtin(__builtin_amdgcn_global_load_async_to_lds_b128)
#  define HAVE_ASYNC_LDS 1
# endif
#endif
#ifndef HAVE_ASYNC_LDS
# define HAVE_ASYNC_LDS 0
#endif
#if defined(__has_builtin)
# if __has_builtin(__builtin_amdgcn_s_wait_asynccnt)
#  define HAVE_WAIT_ASYNC 1
# endif
#endif
#ifndef HAVE_WAIT_ASYNC
# define HAVE_WAIT_ASYNC 0
#endif

// ---- WMMA operand loaders (layouts per CDNA5 ISA 7.12.2, 16-bit A/B) -------
// A-tile (16x32): lanes 0-15: row=m0+lane,   K=k0+{0..7,16..23}
//                 lanes 16-31: row=m0+l-16,  K=k0+{8..15,24..31}
__device__ __forceinline__ v16bf load_a_tile(const bf16_t* __restrict__ base,
                                             int ld, int m0, int k0, int lane) {
  const int r  = m0 + (lane & 15);
  const int kk = k0 + ((lane >> 4) << 3);
  const bf16_t* p = base + (long)r * ld + kk;
  v8bf lo = *reinterpret_cast<const v8bf*>(p);
  v8bf hi = *reinterpret_cast<const v8bf*>(p + 16);
  return __builtin_shufflevector(lo, hi, 0,1,2,3,4,5,6,7,8,9,10,11,12,13,14,15);
}

// B-tile (32x16), W [N,K] row-major; B column n == W row n0+n.
//   lanes 0-15: col=n0+lane, K=k0+{0..7,16..23}; lanes 16-31: K=k0+{8..15,24..31}
__device__ __forceinline__ v16bf load_b_tile(const bf16_t* __restrict__ base,
                                             int ld, int n0, int k0, int lane) {
  const int r  = n0 + (lane & 15);
  const int kk = k0 + ((lane >> 4) << 3);
  const bf16_t* p = base + (long)r * ld + kk;
  v8bf lo = *reinterpret_cast<const v8bf*>(p);
  v8bf hi = *reinterpret_cast<const v8bf*>(p + 16);
  return __builtin_shufflevector(lo, hi, 0,1,2,3,4,5,6,7,8,9,10,11,12,13,14,15);
}

__device__ __forceinline__ v8f wmma_bf16(v16bf a, v16bf b, v8f c) {
  return __builtin_amdgcn_wmma_f32_16x16x32_bf16(false, a, false, b,
                                                 (short)0, c, false, false);
}

__device__ __forceinline__ float sigmoidf_(float x) {
  return 1.0f / (1.0f + __expf(-x));
}

__device__ __forceinline__ void async_copy16(const void* g, void* l) {
#if HAVE_ASYNC_LDS
  __builtin_amdgcn_global_load_async_to_lds_b128(
      (as1_v4ip)(as1_vp)(void*)g,          // global src (cast away const)
      (as3_v4ip)(as3_vp)l,                 // LDS dst
      0, 0);                               // imm offset, imm cpol
#else
  *reinterpret_cast<v8bf*>(l) = *reinterpret_cast<const v8bf*>(g);
#endif
}

__device__ __forceinline__ void wait_async_and_barrier() {
#if HAVE_ASYNC_LDS
# if HAVE_WAIT_ASYNC
  __builtin_amdgcn_s_wait_asynccnt(0);
# else
  asm volatile("s_wait_asynccnt 0x0" ::: "memory");
# endif
#endif
  __syncthreads();
}

// ---- Block GEMM: C = A[M,K] * W[N,K]^T + bias ------------------------------
// 512 threads = 16 waves as 4(mt) x 4(ng); macro-tile 64x256; K chunks of 32
// double-buffered in LDS via async global->LDS copies (16B per lane).
// out_mode 0: C[row*ldc+n];  out_mode 1: row=t*32+b -> C[(b*64+t+1)*32000+n]
#define BLK_M 64
#define BLK_N 256

__global__ __launch_bounds__(512) void gemm_block_kernel(
    const bf16_t* __restrict__ A, int lda,
    const bf16_t* __restrict__ W, int ldw,
    const float* __restrict__ bias,
    float* __restrict__ C, int M, int N, int K, int ldc, int out_mode)
{
  __shared__ bf16_t smA[2][BLK_M * 32];   //  8 KB
  __shared__ bf16_t smB[2][BLK_N * 32];   // 32 KB

  const int lane = threadIdx.x & 31;
  const int wv   = threadIdx.x >> 5;        // 0..15
  const int m0l  = (wv >> 2) << 4;          // 0,16,32,48
  const int n0l  = (wv & 3)  << 6;          // 0,64,128,192
  const int m0b  = blockIdx.y * BLK_M;
  const int n0b  = blockIdx.x * BLK_N;

  // Stage one 32-wide K chunk of the 64x256 macro-tile: 1280 x 16B segments.
  auto stage = [&](int buf, int k0) {
    for (int s = threadIdx.x; s < 1280; s += 512) {
      const bf16_t* g;
      bf16_t* l;
      if (s < 256) {                          // A: 64 rows x 32 cols
        const int r = s >> 2, c = (s & 3) << 3;
        int gr = m0b + r;
        if (gr >= M) gr = M - 1;              // clamp tail rows (stores guarded)
        g = A + (long)gr * lda + k0 + c;
        l = &smA[buf][r * 32 + c];
      } else {                                // B: 256 rows x 32 cols
        const int ss = s - 256;
        const int r = ss >> 2, c = (ss & 3) << 3;
        g = W + (long)(n0b + r) * ldw + k0 + c;
        l = &smB[buf][r * 32 + c];
      }
      async_copy16(g, l);
    }
  };

  stage(0, 0);
  wait_async_and_barrier();

  v8f acc[4] = {};
  int cur = 0;
  for (int k0 = 0; k0 < K; k0 += 32) {
    const int nxt = cur ^ 1;
    const bool more = (k0 + 32) < K;
    if (more) stage(nxt, k0 + 32);            // prefetch next chunk (async)
    const v16bf a = load_a_tile(smA[cur], 32, m0l, 0, lane);
#pragma unroll
    for (int t = 0; t < 4; ++t) {
      const v16bf b = load_b_tile(smB[cur], 32, n0l + (t << 4), 0, lane);
      acc[t] = wmma_bf16(a, b, acc[t]);
    }
    if (more) wait_async_and_barrier();
    cur = nxt;
  }

  const int nn = lane & 15, rb = (lane >> 4) << 3;
#pragma unroll
  for (int t = 0; t < 4; ++t) {
    const int ncol = n0b + n0l + (t << 4) + nn;
    const float bv = bias ? bias[ncol] : 0.0f;
#pragma unroll
    for (int v = 0; v < 8; ++v) {
      const int row = m0b + m0l + rb + v;
      if (row < M) {
        const float val = acc[t][v] + bv;
        if (out_mode == 0) {
          C[(long)row * ldc + ncol] = val;
        } else {
          const int tt = row >> 5, bb = row & 31;    // row = t*32 + b
          C[((long)bb * 64 + tt + 1) * 32000 + ncol] = val;
        }
      }
    }
  }
}

// ---- Fused LSTM step: gates = pre[t] + h @ Whh^T ; c,h update --------------
// grid.y = direction (encoder fwd/bwd fused; decoder uses grid.y==1).
// Hd = per-gate hidden width (== K); HLD = row stride (1024 packs enc dirs).
__global__ __launch_bounds__(128) void lstm_step_kernel(
    const float*  __restrict__ pre0, const float*  __restrict__ pre1,
    const bf16_t* __restrict__ Whh0, const bf16_t* __restrict__ Whh1,
    const bf16_t* __restrict__ hin,        // [32, HLD]
    bf16_t*       __restrict__ hout,       // [32, HLD]
    float*        __restrict__ cst,        // [32, HLD], in place
    int Hd, int HLD)
{
  const int dir = blockIdx.y;
  const float*  pre = dir ? pre1 : pre0;
  const bf16_t* Whh = dir ? Whh1 : Whh0;
  const int dirOff = dir * Hd;

  const int lane = threadIdx.x & 31;
  const int wid  = (blockIdx.x * blockDim.x + threadIdx.x) >> 5;
  const int qTiles = Hd >> 4;
  const int mt = wid / qTiles;
  const int q  = wid - mt * qTiles;
  if (mt >= 2) return;                     // B=32 -> 2 row tiles
  const int m0 = mt << 4;
  const bf16_t* hbase = hin + dirOff;

  v8f acc[4] = {};                         // i, f, g, o tiles, same columns
  for (int k0 = 0; k0 < Hd; k0 += 32) {
    const v16bf a = load_a_tile(hbase, HLD, m0, k0, lane);
#pragma unroll
    for (int g = 0; g < 4; ++g) {
      const v16bf b = load_b_tile(Whh, Hd, g * Hd + (q << 4), k0, lane);
      acc[g] = wmma_bf16(a, b, acc[g]);
    }
  }

  const int nn = lane & 15, rb = (lane >> 4) << 3;
  const int col = (q << 4) + nn;
  const long G = 4L * Hd;
#pragma unroll
  for (int v = 0; v < 8; ++v) {
    const int row = m0 + rb + v;
    const float gi = acc[0][v] + pre[row * G + 0 * Hd + col];
    const float gf = acc[1][v] + pre[row * G + 1 * Hd + col];
    const float gg = acc[2][v] + pre[row * G + 2 * Hd + col];
    const float go = acc[3][v] + pre[row * G + 3 * Hd + col];
    const long ci = (long)row * HLD + dirOff + col;
    const float cn = sigmoidf_(gf) * cst[ci] + sigmoidf_(gi) * tanhf(gg);
    cst[ci] = cn;
    hout[ci] = (bf16_t)(sigmoidf_(go) * tanhf(cn));
  }
}

// ---- Small helpers ----------------------------------------------------------
__global__ void gather_emb_kernel(const int* __restrict__ tok, int tokStride,
                                  const float* __restrict__ emb,
                                  bf16_t* __restrict__ out) {
  const int m = blockIdx.x;                 // m = t*32 + b
  const int t = m >> 5, b = m & 31;
  const int tk = tok[b * tokStride + t];
  const float* s = emb + (long)tk * 256;
  bf16_t* d = out + (long)m * 256;
  d[threadIdx.x] = (bf16_t)s[threadIdx.x];  // blockDim.x == 256 == E
}

__global__ void cvt_f32_bf16_kernel(const float* __restrict__ in,
                                    bf16_t* __restrict__ out, long n) {
  long i = (long)blockIdx.x * blockDim.x + threadIdx.x;
  const long stride = (long)gridDim.x * blockDim.x;
  for (; i < n; i += stride) out[i] = (bf16_t)in[i];
}

__global__ void zero_f32_kernel(float* __restrict__ p, long n) {
  long i = (long)blockIdx.x * blockDim.x + threadIdx.x;
  const long stride = (long)gridDim.x * blockDim.x;
  for (; i < n; i += stride) p[i] = 0.0f;
}

__global__ void zero_out_t0_kernel(float* __restrict__ out) {
  const int b = blockIdx.y;
  const long i = (long)blockIdx.x * blockDim.x + threadIdx.x;
  if (i < 32000) out[(long)b * 64 * 32000 + i] = 0.0f;   // out[b][0][:]
}

// ---------------------------------------------------------------------------
extern "C" void kernel_launch(void* const* d_in, const int* in_sizes, int n_in,
                              void* d_out, int out_size, void* d_ws, size_t ws_size,
                              hipStream_t stream) {
  (void)in_sizes; (void)n_in; (void)out_size; (void)ws_size;

  const int*   src      = (const int*)  d_in[0];
  const int*   trg      = (const int*)  d_in[1];
  const float* enc_emb  = (const float*)d_in[2];
  const float* dec_emb  = (const float*)d_in[3];
  const float* eWihF    = (const float*)d_in[4];
  const float* eWhhF    = (const float*)d_in[5];
  const float* ebF      = (const float*)d_in[6];
  const float* eWihB    = (const float*)d_in[7];
  const float* eWhhB    = (const float*)d_in[8];
  const float* ebB      = (const float*)d_in[9];
  const float* dWih     = (const float*)d_in[10];
  const float* dWhh     = (const float*)d_in[11];
  const float* db       = (const float*)d_in[12];
  const float* fcW      = (const float*)d_in[13];
  const float* fcb      = (const float*)d_in[14];
  float* out = (float*)d_out;

  // ---- workspace carve-up (256B aligned) ----
  char* wsp = (char*)d_ws;
  auto carve = [&](size_t bytes) -> void* {
    void* p = (void*)wsp;
    wsp += (bytes + 255) & ~(size_t)255;
    return p;
  };
  bf16_t* wWihF   = (bf16_t*)carve(2048L * 256 * 2);
  bf16_t* wWhhF   = (bf16_t*)carve(2048L * 512 * 2);
  bf16_t* wWihB   = (bf16_t*)carve(2048L * 256 * 2);
  bf16_t* wWhhB   = (bf16_t*)carve(2048L * 512 * 2);
  bf16_t* wDecWih = (bf16_t*)carve(4096L * 256 * 2);
  bf16_t* wDecWhh = (bf16_t*)carve(4096L * 1024 * 2);
  bf16_t* wFc     = (bf16_t*)carve(32000L * 1024 * 2);
  bf16_t* xsE     = (bf16_t*)carve(4096L * 256 * 2);    // [S*B, E] t-major
  bf16_t* xsD     = (bf16_t*)carve(2016L * 256 * 2);    // [(T-1)*B, E]
  float*  preF    = (float*) carve(4096L * 2048 * 4);   // x@WihF^T + bF
  float*  preB    = (float*) carve(4096L * 2048 * 4);
  float*  preD    = (float*) carve(2016L * 4096 * 4);
  bf16_t* hPing   = (bf16_t*)carve(2L * 32 * 1024 * 2); // ping-pong h (enc)
  float*  cEnc    = (float*) carve(32L * 1024 * 4);     // c (enc concat -> dec c0)
  bf16_t* hsD     = (bf16_t*)carve(63L * 32 * 1024 * 2);// decoder h chain == hs

  // ---- weights -> bf16 ----
  auto cvt = [&](const float* s, bf16_t* d, long n) {
    cvt_f32_bf16_kernel<<<512, 256, 0, stream>>>(s, d, n);
  };
  cvt(eWihF, wWihF, 2048L * 256);
  cvt(eWhhF, wWhhF, 2048L * 512);
  cvt(eWihB, wWihB, 2048L * 256);
  cvt(eWhhB, wWhhB, 2048L * 512);
  cvt(dWih,  wDecWih, 4096L * 256);
  cvt(dWhh,  wDecWhh, 4096L * 1024);
  cvt(fcW,   wFc,   32000L * 1024);

  // ---- embedding gathers (fp32 -> bf16 rows) ----
  gather_emb_kernel<<<128 * 32, 256, 0, stream>>>(src, 128, enc_emb, xsE);
  gather_emb_kernel<<<63 * 32, 256, 0, stream>>>(trg, 64, dec_emb, xsD);

  // ---- zero initial states + out[:,0,:] ----
  zero_f32_kernel<<<64, 256, 0, stream>>>((float*)hPing, 2L * 32 * 1024 / 2);
  zero_f32_kernel<<<64, 256, 0, stream>>>(cEnc, 32L * 1024);
  zero_out_t0_kernel<<<dim3((32000 + 255) / 256, 32), 256, 0, stream>>>(out);

  // ---- block GEMMs (async-LDS double-buffered WMMA) ----
  auto gemm = [&](const bf16_t* A, int lda, const bf16_t* W, int ldw,
                  const float* bias, float* C, int M, int N, int K, int ldc,
                  int mode) {
    dim3 grid(N / BLK_N, (M + BLK_M - 1) / BLK_M);
    gemm_block_kernel<<<grid, 512, 0, stream>>>(A, lda, W, ldw, bias, C,
                                                M, N, K, ldc, mode);
  };

  // hoisted input GEMMs: pre = x @ Wih^T + b
  gemm(xsE, 256, wWihF,   256, ebF, preF, 4096, 2048, 256, 2048, 0);
  gemm(xsE, 256, wWihB,   256, ebB, preB, 4096, 2048, 256, 2048, 0);
  gemm(xsD, 256, wDecWih, 256, db,  preD, 2016, 4096, 256, 4096, 0);

  // ---- encoder: 128 fused bidirectional steps (grid.y = direction) ----
  for (int s = 0; s < 128; ++s) {
    const float*  pf = preF + (long)s * 32 * 2048;
    const float*  pb = preB + (long)(127 - s) * 32 * 2048;   // xs[::-1]
    const bf16_t* hi = hPing + (long)(s & 1) * 32 * 1024;
    bf16_t*       ho = hPing + (long)((s + 1) & 1) * 32 * 1024;
    lstm_step_kernel<<<dim3(16, 2), 128, 0, stream>>>(
        pf, pb, wWhhF, wWhhB, hi, ho, cEnc, 512, 1024);
  }
  // final h (bf16, [hf|hb] packed) lands in hPing[0]; c0 == cEnc.

  // ---- decoder: 63 steps, h chain written straight into hs matrix ----
  for (int t = 0; t < 63; ++t) {
    const bf16_t* hi = (t == 0) ? hPing : hsD + (long)(t - 1) * 32 * 1024;
    bf16_t*       ho = hsD + (long)t * 32 * 1024;
    lstm_step_kernel<<<dim3(32, 1), 128, 0, stream>>>(
        preD + (long)t * 32 * 4096, nullptr, wDecWhh, nullptr,
        hi, ho, cEnc, 1024, 1024);
  }

  // ---- logits: hs[2016,1024] @ fc_W^T + fc_b -> out[b][t+1][:] ----
  gemm(hsD, 1024, wFc, 1024, fcb, out, 2016, 32000, 1024, 0, 1);
}
